// EdgeConvNet_87308095193264
// MI455X (gfx1250) — matrix operand order repaired and checked by reading, not compiled, not run on hardware
//
#include <hip/hip_runtime.h>
#include <math.h>

typedef unsigned short u16;
typedef __attribute__((ext_vector_type(16))) __bf16 v16bf;
typedef __attribute__((ext_vector_type(8)))  float  v8f;

#define BN_EPS 1e-5f

struct U32x8 { uint4 lo, hi; };

__device__ __forceinline__ v16bf frag_cast(uint4 lo, uint4 hi) {
  U32x8 u; u.lo = lo; u.hi = hi;
  return __builtin_bit_cast(v16bf, u);
}

__device__ __forceinline__ u16 f2bf(float f) {
  unsigned u = __builtin_bit_cast(unsigned, f);
  unsigned r = u + 0x7FFFu + ((u >> 16) & 1u);   // round-to-nearest-even
  return (u16)(r >> 16);
}
__device__ __forceinline__ float bf2f(u16 h) {
  return __builtin_bit_cast(float, ((unsigned)h) << 16);
}

// CDNA5 native no-return float atomic max (segment_max aggregation)
__device__ __forceinline__ void gmax_f32(float* p, float v) {
  asm volatile("global_atomic_max_num_f32 %0, %1, off" :: "v"(p), "v"(v) : "memory");
}

// ---------------------------------------------------------------- utilities
__global__ void __launch_bounds__(256) k_fill(float* p, float v, int n) {
  int i = blockIdx.x * 256 + threadIdx.x;
  if (i < n) p[i] = v;
}

__global__ void __launch_bounds__(256) k_fix(float* h, int n) {
  int i = blockIdx.x * 256 + threadIdx.x;
  if (i < n) { float v = h[i]; if (__builtin_isinf(v)) h[i] = 0.f; }
}

__global__ void __launch_bounds__(256) k_fixtobf(float* h, u16* Hb, int n) {
  int i = blockIdx.x * 256 + threadIdx.x;
  if (i < n) { float v = h[i]; if (__builtin_isinf(v)) v = 0.f; h[i] = v; Hb[i] = f2bf(v); }
}

// ------------------------------------------- edge feature builders (gather)
// conv1 input: [x_dst, x_src - x_dst] (E x 32), also accumulate BN0 stats
__global__ void __launch_bounds__(256)
k_gather1(const float* __restrict__ x, const int* __restrict__ src,
          const int* __restrict__ dst, u16* __restrict__ A,
          float* __restrict__ stats, int E) {
  __shared__ float sSum[32];
  __shared__ float sSq[32];
  int tid = threadIdx.x;
  if (tid < 32) { sSum[tid] = 0.f; sSq[tid] = 0.f; }
  __syncthreads();
  long gid = (long)blockIdx.x * 256 + tid;
  int e = (int)(gid >> 5);
  int c = (int)(gid & 31);
  float v = 0.f;
  bool ok = (e < E);
  if (ok) {
    int d = dst[e], s = src[e];
    v = (c < 16) ? x[d * 16 + c] : (x[s * 16 + (c - 16)] - x[d * 16 + (c - 16)]);
    A[(long)e * 32 + c] = f2bf(v);
  }
  atomicAdd(&sSum[c], ok ? v : 0.f);
  atomicAdd(&sSq[c],  ok ? v * v : 0.f);
  __syncthreads();
  if (tid < 32) { atomicAdd(&stats[tid], sSum[tid]); atomicAdd(&stats[32 + tid], sSq[tid]); }
}

// conv2 input: [h_dst, h_src - h_dst] (E x 256), h is N x 128 f32
__global__ void __launch_bounds__(256)
k_gather2(const float* __restrict__ h, const int* __restrict__ src,
          const int* __restrict__ dst, u16* __restrict__ A, int E) {
  long gid = (long)blockIdx.x * 256 + threadIdx.x;
  int e = (int)(gid >> 8), c = (int)(gid & 255);
  if (e >= E) return;
  int d = dst[e], s = src[e];
  float v = (c < 128) ? h[d * 128 + c] : (h[s * 128 + (c - 128)] - h[d * 128 + (c - 128)]);
  A[gid] = f2bf(v);
}

// ------------------------------------------- BN-fold + transpose weight prep
__global__ void __launch_bounds__(256)
k_prep(const float* __restrict__ W, const float* __restrict__ c,
       const float* __restrict__ g, const float* __restrict__ b,
       const float* __restrict__ stats, float invE,
       u16* __restrict__ Wt, float* __restrict__ bias, int din, int dout) {
  int j = threadIdx.x;
  if (j >= dout) return;
  float sh = 0.f;
  for (int k = 0; k < din; ++k) {
    float w = W[(long)k * dout + j];
    float scale = 1.f, shift = 0.f;
    if (g) {
      float mean = stats[k] * invE;
      float var  = stats[din + k] * invE - mean * mean;
      float s    = rsqrtf(var + BN_EPS);
      scale = g[k] * s;
      shift = b[k] - mean * scale;
    }
    Wt[(long)j * din + k] = f2bf(w * scale);
    sh += shift * w;
  }
  bias[j] = c[j] + sh;
}

// ------------------------------------------- WMMA bf16 GEMM + bias/ReLU/stats
// A: rows x DIN bf16 row-major. Wt: DOUT x DIN bf16 (transposed weights).
// Out: rows x DOUT bf16. stats (optional): [sum(DOUT), sumsq(DOUT)] atomics.
// Block = 8 waves; each wave owns 32 rows (2 row-tiles) x all DOUT columns,
// so each LDS B-fragment feeds two WMMA ops. Weight K-slices double-buffered.
template <int DIN, int DOUT>
__global__ void __launch_bounds__(256)
k_gemm(const u16* __restrict__ A, const u16* __restrict__ Wt,
       const float* __restrict__ bias, u16* __restrict__ Out,
       float* __restrict__ stats, int rows, int relu) {
  constexpr int NT = DOUT / 16;
  constexpr int NCHUNK = DIN / 32;
  __shared__ __align__(16) u16 sW[2][DOUT * 32];
  __shared__ float sSum[DOUT];
  __shared__ float sSq[DOUT];
  int tid = threadIdx.x;
  if (stats && tid < DOUT) { sSum[tid] = 0.f; sSq[tid] = 0.f; }
  int wave = tid >> 5, lane = tid & 31;
  int r16 = lane & 15, half = lane >> 4;
  int rowbase = blockIdx.x * 256 + wave * 32;
  int arow0 = rowbase + r16;
  int arow1 = rowbase + 16 + r16;
  if (arow0 >= rows) arow0 = rows - 1;     // clamp (stores are guarded)
  if (arow1 >= rows) arow1 = rows - 1;
  const u16* Arow0 = A + (long)arow0 * DIN;
  const u16* Arow1 = A + (long)arow1 * DIN;

  v8f acc0[NT], acc1[NT];
#pragma unroll
  for (int t = 0; t < NT; ++t) {
    acc0[t] = (v8f){0.f, 0.f, 0.f, 0.f, 0.f, 0.f, 0.f, 0.f};
    acc1[t] = (v8f){0.f, 0.f, 0.f, 0.f, 0.f, 0.f, 0.f, 0.f};
  }

  auto stage = [&](int k, int p) {
#pragma unroll
    for (int i = tid; i < DOUT * 4; i += 256) {
      int col = i >> 2, kk = (i & 3) * 8;
      *(uint4*)(sW[p] + col * 32 + kk) = *(const uint4*)(Wt + (long)col * DIN + k + kk);
    }
  };
  stage(0, 0);
  int p = 0;
  for (int cch = 0; cch < NCHUNK; ++cch) {
    int k = cch * 32;
    __syncthreads();
    if (cch + 1 < NCHUNK) stage(k + 32, p ^ 1);
    // A fragments per ISA layout: lanes 0-15 -> K k..k+7 / k+16..k+23,
    //                             lanes 16-31 -> K k+8..k+15 / k+24..k+31
    v16bf a0 = frag_cast(*(const uint4*)(Arow0 + k + half * 8),
                         *(const uint4*)(Arow0 + k + 16 + half * 8));
    v16bf a1 = frag_cast(*(const uint4*)(Arow1 + k + half * 8),
                         *(const uint4*)(Arow1 + k + 16 + half * 8));
#pragma unroll
    for (int t = 0; t < NT; ++t) {
      // B fragment: lane holds column (t*16+r16), K half*8..+7 / 16+half*8..+7
      const u16* wp = sW[p] + (t * 16 + r16) * 32 + half * 8;
      v16bf bf = frag_cast(*(const uint4*)wp, *(const uint4*)(wp + 16));
      acc0[t] = __builtin_amdgcn_wmma_f32_16x16x32_bf16(
          false, a0, false, bf, (short)0, acc0[t], false, false);
      acc1[t] = __builtin_amdgcn_wmma_f32_16x16x32_bf16(
          false, a1, false, bf, (short)0, acc1[t], false, false);
    }
    p ^= 1;
  }

  bool full = (rowbase + 32) <= rows;       // uniform within wave
#pragma unroll
  for (int t = 0; t < NT; ++t) {
    int col = t * 16 + r16;
    float bv = bias[col];
    float s1 = 0.f, s2 = 0.f;
#pragma unroll
    for (int r = 0; r < 8; ++r) {
      float v0 = acc0[t][r] + bv;
      float v1 = acc1[t][r] + bv;
      if (relu) { v0 = fmaxf(v0, 0.f); v1 = fmaxf(v1, 0.f); }
      int row0 = rowbase + r + 8 * half;     // C/D layout: M = r + 8*half
      int row1 = row0 + 16;
      if (full) {                            // fast path: unguarded stores
        Out[(long)row0 * DOUT + col] = f2bf(v0);
        Out[(long)row1 * DOUT + col] = f2bf(v1);
        s1 += v0 + v1; s2 += v0 * v0 + v1 * v1;
      } else {
        if (row0 < rows) { Out[(long)row0 * DOUT + col] = f2bf(v0); s1 += v0; s2 += v0 * v0; }
        if (row1 < rows) { Out[(long)row1 * DOUT + col] = f2bf(v1); s1 += v1; s2 += v1 * v1; }
      }
    }
    if (stats) { atomicAdd(&sSum[col], s1); atomicAdd(&sSq[col], s2); }
  }
  if (stats) {
    __syncthreads();
    if (tid < DOUT) { atomicAdd(&stats[tid], sSum[tid]); atomicAdd(&stats[DOUT + tid], sSq[tid]); }
  }
}

// ------------------------------------------- final BN + segment_max (atomic)
template <int DOUT>
__global__ void __launch_bounds__(256)
k_aggregate(const u16* __restrict__ A, const float* __restrict__ stats,
            const float* __restrict__ g, const float* __restrict__ b,
            const int* __restrict__ dst, float* __restrict__ h,
            float invE, int E) {
  long gid = (long)blockIdx.x * 256 + threadIdx.x;
  int e = (int)(gid / DOUT), c = (int)(gid % DOUT);
  if (e >= E) return;
  float mean = stats[c] * invE;
  float var  = stats[DOUT + c] * invE - mean * mean;
  float s    = rsqrtf(var + BN_EPS) * g[c];
  float val  = (bf2f(A[gid]) - mean) * s + b[c];
  gmax_f32(&h[(long)dst[e] * DOUT + c], val);
}

// ------------------------------------------- head tail: (256x1) dot + sigmoid
__global__ void __launch_bounds__(256)
k_final(const u16* __restrict__ H, const float* __restrict__ W4,
        const float* __restrict__ c4, float* __restrict__ out, int N) {
  int n = blockIdx.x * 256 + threadIdx.x;
  if (n >= N) return;
  float acc = c4[0];
  const u16* row = H + (long)n * 256;
  for (int k = 0; k < 256; ++k) acc += bf2f(row[k]) * W4[k];
  out[n] = 1.f / (1.f + expf(-acc));
}

// =========================================================================
extern "C" void kernel_launch(void* const* d_in, const int* in_sizes, int n_in,
                              void* d_out, int out_size, void* d_ws, size_t ws_size,
                              hipStream_t stream) {
  const long N = in_sizes[0] / 16;
  const long E = in_sizes[1] / 2;
  const long Epad = ((E + 255) / 256) * 256;
  const float invE = 1.f / (float)E;

  const float* x  = (const float*)d_in[0];
  const int* ei   = (const int*)d_in[1];
  const int* src  = ei;
  const int* dstp = ei + E;
  auto P = [&](int i) { return (const float*)d_in[i]; };

  char* w = (char*)d_ws;
  size_t off = 0;
  auto alloc = [&](size_t bytes) -> char* {
    char* p = w + off;
    off = (off + bytes + 255) & ~(size_t)255;
    return p;
  };
  u16*   Abuf  = (u16*)alloc(Epad * 256 * 2);
  u16*   Bbuf  = (u16*)alloc(Epad * 256 * 2);
  float* h1    = (float*)alloc(N * 128 * 4);
  float* h2    = (float*)alloc(N * 256 * 4);
  u16*   Hb    = (u16*)alloc(N * 256 * 2);
  u16*   Hb2   = (u16*)alloc(N * 256 * 2);
  float* st    = (float*)alloc(16 * 512 * 4);
  u16*   Wt0   = (u16*)alloc(256 * 256 * 2);
  u16*   Wt1   = (u16*)alloc(256 * 256 * 2);
  float* bias0 = (float*)alloc(256 * 4);
  float* bias1 = (float*)alloc(256 * 4);

  const unsigned T = 256;
  auto blk  = [](long n) { return dim3((unsigned)((n + 255) / 256)); };
  unsigned gE = (unsigned)((E + 255) / 256);   // 256 rows per block
  unsigned gN = (unsigned)((N + 255) / 256);

  // init: node maxima to -inf, stats to 0
  k_fill<<<blk(N * 128), T, 0, stream>>>(h1, -__builtin_inff(), (int)(N * 128));
  k_fill<<<blk(N * 256), T, 0, stream>>>(h2, -__builtin_inff(), (int)(N * 256));
  k_fill<<<blk(16 * 512), T, 0, stream>>>(st, 0.f, 16 * 512);

  // ---------------- EdgeConv 1 (mlp1: BN0 folded into W1; BN1/BN2 folded fwd)
  k_gather1<<<blk(E * 32), T, 0, stream>>>(x, src, dstp, Abuf, st + 0 * 512, (int)E);
  k_prep<<<1, T, 0, stream>>>(P(4),  P(5),  P(2),  P(3),  st + 0 * 512, invE, Wt0, bias0, 32, 128);
  k_gemm<32, 128><<<gE, T, 0, stream>>>(Abuf, Wt0, bias0, Bbuf, st + 1 * 512, (int)E, 1);
  k_prep<<<1, T, 0, stream>>>(P(8),  P(9),  P(6),  P(7),  st + 1 * 512, invE, Wt1, bias1, 128, 128);
  k_gemm<128, 128><<<gE, T, 0, stream>>>(Bbuf, Wt1, bias1, Abuf, st + 2 * 512, (int)E, 1);
  k_prep<<<1, T, 0, stream>>>(P(12), P(13), P(10), P(11), st + 2 * 512, invE, Wt0, bias0, 128, 128);
  k_gemm<128, 128><<<gE, T, 0, stream>>>(Abuf, Wt0, bias0, Bbuf, st + 3 * 512, (int)E, 1);
  k_aggregate<128><<<blk(E * 128), T, 0, stream>>>(Bbuf, st + 3 * 512, P(14), P(15), dstp, h1, invE, (int)E);
  k_fix<<<blk(N * 128), T, 0, stream>>>(h1, (int)(N * 128));

  // ---------------- EdgeConv 2 (mlp2)
  k_gather2<<<blk(E * 256), T, 0, stream>>>(h1, src, dstp, Abuf, (int)E);
  k_prep<<<1, T, 0, stream>>>(P(16), P(17), nullptr, nullptr, nullptr, invE, Wt0, bias0, 256, 256);
  k_gemm<256, 256><<<gE, T, 0, stream>>>(Abuf, Wt0, bias0, Bbuf, st + 4 * 512, (int)E, 1);
  k_prep<<<1, T, 0, stream>>>(P(20), P(21), P(18), P(19), st + 4 * 512, invE, Wt1, bias1, 256, 256);
  k_gemm<256, 256><<<gE, T, 0, stream>>>(Bbuf, Wt1, bias1, Abuf, st + 5 * 512, (int)E, 1);
  k_prep<<<1, T, 0, stream>>>(P(24), P(25), P(22), P(23), st + 5 * 512, invE, Wt0, bias0, 256, 256);
  k_gemm<256, 256><<<gE, T, 0, stream>>>(Abuf, Wt0, bias0, Bbuf, st + 6 * 512, (int)E, 1);
  k_aggregate<256><<<blk(E * 256), T, 0, stream>>>(Bbuf, st + 6 * 512, P(26), P(27), dstp, h2, invE, (int)E);
  k_fixtobf<<<blk(N * 256), T, 0, stream>>>(h2, Hb, (int)(N * 256));

  // ---------------- Head (node-level)
  k_prep<<<1, T, 0, stream>>>(P(28), P(29), nullptr, nullptr, nullptr, 0.f, Wt0, bias0, 256, 256);
  k_gemm<256, 256><<<gN, T, 0, stream>>>(Hb, Wt0, bias0, Hb2, nullptr, (int)N, 1);
  k_prep<<<1, T, 0, stream>>>(P(30), P(31), nullptr, nullptr, nullptr, 0.f, Wt1, bias1, 256, 256);
  k_gemm<256, 256><<<gN, T, 0, stream>>>(Hb2, Wt1, bias1, Hb, nullptr, (int)N, 1);
  k_prep<<<1, T, 0, stream>>>(P(32), P(33), nullptr, nullptr, nullptr, 0.f, Wt0, bias0, 256, 256);
  k_gemm<256, 256><<<gN, T, 0, stream>>>(Hb, Wt0, bias0, Hb2, nullptr, (int)N, 0);
  k_final<<<blk(N), T, 0, stream>>>(Hb2, P(34), P(35), (float*)d_out, (int)N);
}
